// iLoRAModel_68891275427994
// MI455X (gfx1250) — compile-verified, matrix operand order checked by reading, not run on hardware
//
#include <hip/hip_runtime.h>
#include <hip/hip_bf16.h>

// ---------------- model dims ----------------
constexpr int BB   = 4;      // batch
constexpr int SS   = 128;    // seq len
constexpr int DD   = 768;    // model dim
constexpr int NL   = 2;      // layers
constexpr int NH   = 12;     // heads
constexpr int NE   = 4;      // experts
constexpr int RRK  = 8;      // lora rank
constexpr int DHH  = 64;     // head dim
constexpr int DFF  = 3072;   // ffn dim
constexpr int HID  = 128;    // gate hidden
constexpr int NOUT = 20001;  // vocab out (NI+1)
constexpr int MROWS = NE * BB * SS;   // 2048 batched rows (experts folded into M)
constexpr float LSCALE = 16.0f / 8.0f; // lora alpha/r

typedef __attribute__((ext_vector_type(8)))  _Float16 v8h;
typedef __attribute__((ext_vector_type(16))) _Float16 v16h;
typedef __attribute__((ext_vector_type(8)))  float    v8f;

// Load one 16x32 f16 WMMA fragment row from LDS.
// Per ISA 7.12.2: lane (l&15) = matrix row, s = l>>4 selects K-halves:
//   frag halves 0..7  = K[s*8 .. s*8+7]
//   frag halves 8..15 = K[16+s*8 .. 23+s*8]
// => two aligned 16B LDS loads per fragment (ds_load_b128).
static __device__ inline v16h ld_frag(const _Float16* rowk, int s) {
  v8h lo = *reinterpret_cast<const v8h*>(rowk + s * 8);
  v8h hi = *reinterpret_cast<const v8h*>(rowk + 16 + s * 8);
  return __builtin_shufflevector(lo, hi, 0,1,2,3,4,5,6,7,8,9,10,11,12,13,14,15);
}

static __device__ inline v8f wmma16(v16h a, v16h b, v8f c) {
  return __builtin_amdgcn_wmma_f32_16x16x32_f16(false, a, false, b, (short)0, c,
                                                false, false);
}

// pack two f32 -> one dword of two f16 (lowers to v_cvt_pk_f16_f32)
static __device__ inline unsigned pk_h2(float a, float b) {
  union { _Float16 h[2]; unsigned u; } p;
  p.h[0] = (_Float16)a; p.h[1] = (_Float16)b;
  return p.u;
}

// branch-free tanh-approx gelu (avoids libm tanhf's branchy expansion)
static __device__ inline float gelu_tanh(float v) {
  float y  = 0.7978845608f * (v + 0.044715f * v * v * v);
  float e2 = __expf(-2.0f * fabsf(y));
  float th = (1.0f - e2) / (1.0f + e2);
  th = (y < 0.0f) ? -th : th;
  return 0.5f * v * (1.0f + th);
}

// CDNA5 async global->LDS copy (16B per lane), tracked by ASYNCcnt.
// VGLOBAL async encoding: VDST = LDS byte address VGPR, VADDR = 64-bit global
// address, no SADDR -> "off".  LDS offset = low 32 bits of generic shared ptr.
static __device__ inline void async_copy_b128(void* ldst, const void* gsrc) {
  unsigned loff = (unsigned)(size_t)ldst;
  asm volatile("global_load_async_to_lds_b128 %0, %1, off"
               :: "v"(loff), "v"(gsrc) : "memory");
}
static __device__ inline void async_wait0() {
  asm volatile("s_wait_asynccnt 0" ::: "memory");
}

// ---------------- gating MLP (tiny) ----------------
__global__ __launch_bounds__(256)
void gate_kernel(const int* __restrict__ seq, const int* __restrict__ slen,
                 const float* __restrict__ item_emb,
                 const float* __restrict__ pW, const float* __restrict__ pb,
                 const float* __restrict__ gW1, const float* __restrict__ gb1,
                 const float* __restrict__ gW2, const float* __restrict__ gb2,
                 float* __restrict__ gate) {
  __shared__ float le[DD];
  __shared__ float hid[HID];
  __shared__ float lg[NE];
  int tid = threadIdx.x;
  for (int b = 0; b < BB; b++) {
    int last = seq[b * 50 + (slen[b] - 1)];
    const float* ie = item_emb + (size_t)last * HID;
    for (int d = tid; d < DD; d += 256) {
      float acc = pb[d];
      for (int k = 0; k < HID; k++) acc += ie[k] * pW[k * DD + d];
      le[d] = acc;
    }
    __syncthreads();
    if (tid < HID) {
      float acc = gb1[tid];
      for (int d = 0; d < DD; d++) acc += le[d] * gW1[d * HID + tid];
      hid[tid] = fmaxf(acc, 0.0f);
    }
    __syncthreads();
    if (tid < NE) {
      float acc = gb2[tid];
      for (int j = 0; j < HID; j++) acc += hid[j] * gW2[j * NE + tid];
      lg[tid] = acc;
    }
    __syncthreads();
    if (tid == 0) {
      float mx = fmaxf(fmaxf(lg[0], lg[1]), fmaxf(lg[2], lg[3]));
      float s0 = __expf(lg[0] - mx), s1 = __expf(lg[1] - mx);
      float s2 = __expf(lg[2] - mx), s3 = __expf(lg[3] - mx);
      float inv = 1.0f / (s0 + s1 + s2 + s3);
      gate[b * NE + 0] = s0 * inv; gate[b * NE + 1] = s1 * inv;
      gate[b * NE + 2] = s2 * inv; gate[b * NE + 3] = s3 * inv;
    }
    __syncthreads();
  }
}

// ---------------- embedding gather (replicated across experts) ----------------
__global__ __launch_bounds__(256)
void embed_kernel(const int* __restrict__ ids, const float* __restrict__ emb,
                  float* __restrict__ x) {
  int row = blockIdx.x;              // 0 .. MROWS-1, expert-major
  int bs  = row % (BB * SS);         // b*SS + s
  int tok = ids[bs];
  for (int d = threadIdx.x; d < DD; d += 256)
    x[(size_t)row * DD + d] = emb[(size_t)tok * DD + d];
}

// ---------------- rmsnorm ----------------
__global__ __launch_bounds__(256)
void rmsnorm_kernel(const float* __restrict__ x, const float* __restrict__ w,
                    float* __restrict__ out) {
  __shared__ float red[256];
  int row = blockIdx.x, tid = threadIdx.x;
  const float* xr = x + (size_t)row * DD;
  float ss = 0.0f;
  for (int d = tid; d < DD; d += 256) { float v = xr[d]; ss += v * v; }
  red[tid] = ss; __syncthreads();
  for (int s = 128; s > 0; s >>= 1) {
    if (tid < s) red[tid] += red[tid + s];
    __syncthreads();
  }
  float inv = rsqrtf(red[0] / (float)DD + 1e-6f);
  for (int d = tid; d < DD; d += 256)
    out[(size_t)row * DD + d] = xr[d] * inv * w[d];
}

// ---------------- generic WMMA GEMM: C = op(A[M,K] @ W[K,N]) ----------------
// 128x128 tile, 8 waves, K-step 32, double-buffered LDS, packed f16 staging.
// flags bit0: gelu epilogue; bit1: accumulate into C (f32); bit2: store f16 to C16
constexpr int LDT = 40;  // LDS row stride in halves (80B, 16B-aligned, 32+8 padded)

__global__ __launch_bounds__(256)
void gemm_kernel(const float* __restrict__ A, const float* __restrict__ W,
                 float* __restrict__ C, _Float16* __restrict__ C16,
                 int M, int N, int K, int flags) {
  __shared__ _Float16 As[2][128 * LDT];  // As[m][k]   (A tile, row-major)
  __shared__ _Float16 Bs[2][128 * LDT];  // Bs[n][k] = W[k][n]  (W tile, transposed)
  const int tid = threadIdx.x;
  const int wave = tid >> 5, lane = tid & 31;
  const int l15 = lane & 15, sh = lane >> 4;
  const int row0 = blockIdx.y * 128, col0 = blockIdx.x * 128;
  const int nk = K >> 5;

  // staging coords: A pairs (m, kp): kp = tid&15, m = tid>>4 + 16*i
  //                 B pairs (n, kp): n = tid&127, kp = tid>>7 + 2*i
  const int a_kp = tid & 15, a_m0 = tid >> 4;
  const int b_n  = tid & 127, b_kp0 = tid >> 7;

  float2 areg[8];
  float  breg[8][2];

  auto ldg = [&](int kt) {
#pragma unroll
    for (int i = 0; i < 8; i++) {
      int m = a_m0 + 16 * i;
      areg[i] = *reinterpret_cast<const float2*>(
          &A[(size_t)(row0 + m) * K + kt + 2 * a_kp]);
    }
#pragma unroll
    for (int i = 0; i < 8; i++) {
      int kp = b_kp0 + 2 * i;
      breg[i][0] = W[(size_t)(kt + 2 * kp)     * N + col0 + b_n];
      breg[i][1] = W[(size_t)(kt + 2 * kp + 1) * N + col0 + b_n];
    }
  };
  auto sts = [&](int buf) {
#pragma unroll
    for (int i = 0; i < 8; i++) {
      int m = a_m0 + 16 * i;
      *reinterpret_cast<unsigned*>(&As[buf][m * LDT + 2 * a_kp]) =
          pk_h2(areg[i].x, areg[i].y);
    }
#pragma unroll
    for (int i = 0; i < 8; i++) {
      int kp = b_kp0 + 2 * i;
      *reinterpret_cast<unsigned*>(&Bs[buf][b_n * LDT + 2 * kp]) =
          pk_h2(breg[i][0], breg[i][1]);
    }
  };

  v8f acc[8] = {};
  ldg(0);
  sts(0);
  __syncthreads();
  for (int t = 0; t < nk; t++) {
    if (t + 1 < nk) ldg((t + 1) << 5);        // HBM loads in flight during WMMA
    const _Float16* as = As[t & 1];
    const _Float16* bs = Bs[t & 1];
    v16h af = ld_frag(&as[(wave * 16 + l15) * LDT], sh);
    v16h bf[8];
#pragma unroll
    for (int nt = 0; nt < 8; nt++)            // preload all B frags (pipelined ds)
      bf[nt] = ld_frag(&bs[(nt * 16 + l15) * LDT], sh);
#pragma unroll
    for (int nt = 0; nt < 8; nt++)            // back-to-back WMMAs
      acc[nt] = wmma16(af, bf[nt], acc[nt]);
    if (t + 1 < nk) sts((t + 1) & 1);
    __syncthreads();
  }
#pragma unroll
  for (int nt = 0; nt < 8; nt++)
#pragma unroll
    for (int r = 0; r < 8; r++) {
      int m = row0 + wave * 16 + sh * 8 + r;   // C layout: VGPR r -> M=r+8*s
      int n = col0 + nt * 16 + l15;            //           lane&15 -> N
      float v = acc[nt][r];
      if (flags & 1) v = gelu_tanh(v);
      size_t idx = (size_t)m * N + n;
      if (flags & 4) {
        C16[idx] = (_Float16)v;                // f16 output (q/k/v path)
      } else {
        if (flags & 2) v += C[idx];
        C[idx] = v;
      }
    }
}

// ---------------- rank-8 LoRA fixup: dst += SCALE*(h@A_e)@B_e (f16 dst) ------
__global__ __launch_bounds__(256)
void lora_kernel(const float* __restrict__ h, const float* __restrict__ Am,
                 const float* __restrict__ Bm, _Float16* __restrict__ dst, int l) {
  int row = blockIdx.x * blockDim.x + threadIdx.x;
  if (row >= MROWS) return;
  int e = row / (BB * SS);
  const float* Ae = Am + ((size_t)(e * NL + l)) * DD * RRK;  // [D,R]
  const float* Be = Bm + ((size_t)(e * NL + l)) * RRK * DD;  // [R,D]
  const float* hr = h + (size_t)row * DD;
  float t[RRK];
#pragma unroll
  for (int r = 0; r < RRK; r++) t[r] = 0.0f;
  for (int d = 0; d < DD; d++) {
    float hv = hr[d];
#pragma unroll
    for (int r = 0; r < RRK; r++) t[r] += hv * Ae[d * RRK + r];
  }
  _Float16* dr = dst + (size_t)row * DD;
  for (int d = 0; d < DD; d++) {
    float a = 0.0f;
#pragma unroll
    for (int r = 0; r < RRK; r++) a += t[r] * Be[r * DD + d];
    dr[d] = (_Float16)((float)dr[d] + LSCALE * a);
  }
}

// ---------------- fused per-head attention (WMMA + f32 softmax) --------------
// q/k/v arrive as f16 -> Q/K tiles stream into LDS via async copies (ASYNCcnt)
// while the VALU stages the V transpose; one s_wait_asynccnt + barrier.
constexpr int QLD = 72;   // 64+8 halves (144B, 16B aligned)
constexpr int PLD = 136;  // 128+8 halves (272B, 16B aligned)

__global__ __launch_bounds__(256)
void attn_kernel(const _Float16* __restrict__ q, const _Float16* __restrict__ k,
                 const _Float16* __restrict__ v, const int* __restrict__ amask,
                 float* __restrict__ o) {
  __shared__ _Float16 Qs[SS * QLD];    // Qs[s][dh]
  __shared__ _Float16 Ks[SS * QLD];    // Ks[s'][dh]  == B-frag rows for Q@K^T
  __shared__ _Float16 Ps[SS * PLD];    // scores -> probs (f16)
  __shared__ _Float16 Vt[DHH * PLD];   // Vt[dh][s'] == B-frag rows for P@V
  int blk = blockIdx.x;
  int e  = blk / (BB * NH);
  int b  = (blk / NH) % BB;
  int hh = blk % NH;
  const size_t rowbase = ((size_t)(e * BB + b)) * SS;
  int tid = threadIdx.x, wave = tid >> 5, lane = tid & 31;
  int l15 = lane & 15, sh = lane >> 4;
#pragma unroll
  for (int i = 0; i < 4; i++) {        // async-stream Q,K head tiles (16B/lane)
    int ee = tid + i * 256;            // 0..1023
    int s2 = ee >> 3, c8 = ee & 7;     // row, 8-half chunk
    size_t g = (rowbase + s2) * DD + hh * DHH + c8 * 8;
    async_copy_b128(&Qs[s2 * QLD + c8 * 8], q + g);
    async_copy_b128(&Ks[s2 * QLD + c8 * 8], k + g);
  }
#pragma unroll
  for (int i = 0; i < 32; i++) {       // stage V transposed (overlaps async)
    int ee = tid + i * 256;
    int s2 = ee >> 6, dh = ee & 63;
    Vt[dh * PLD + s2] = v[(rowbase + s2) * DD + hh * DHH + dh];
  }
  async_wait0();
  __syncthreads();
  v8f acc[8] = {};
#pragma unroll
  for (int kt = 0; kt < DHH; kt += 32) {           // scores = Q @ K^T
    v16h af = ld_frag(&Qs[(wave * 16 + l15) * QLD + kt], sh);
    v16h bf[8];
#pragma unroll
    for (int nt = 0; nt < 8; nt++)
      bf[nt] = ld_frag(&Ks[(nt * 16 + l15) * QLD + kt], sh);
#pragma unroll
    for (int nt = 0; nt < 8; nt++) acc[nt] = wmma16(af, bf[nt], acc[nt]);
  }
#pragma unroll
  for (int nt = 0; nt < 8; nt++)                   // mask + scale -> LDS f16
#pragma unroll
    for (int r = 0; r < 8; r++) {
      int m = wave * 16 + sh * 8 + r;
      int n = nt * 16 + l15;
      float sc = acc[nt][r] * 0.125f;              // 1/sqrt(64)
      bool ok = (n <= m) && (amask[b * SS + n] > 0);
      Ps[m * PLD + n] = (_Float16)(ok ? sc : -30000.0f);
    }
  __syncthreads();
  if (tid < SS) {                                  // f32 softmax per row
    int row = tid;
    float mx = -1e30f;
    for (int j = 0; j < SS; j++) mx = fmaxf(mx, (float)Ps[row * PLD + j]);
    float sum = 0.0f;
    for (int j = 0; j < SS; j++) {
      float ex = __expf((float)Ps[row * PLD + j] - mx);
      sum += ex;
      Ps[row * PLD + j] = (_Float16)ex;
    }
    float inv = 1.0f / sum;
    for (int j = 0; j < SS; j++)
      Ps[row * PLD + j] = (_Float16)((float)Ps[row * PLD + j] * inv);
  }
  __syncthreads();
  v8f acc2[4] = {};
#pragma unroll
  for (int kt = 0; kt < SS; kt += 32) {            // out = P @ V
    v16h af = ld_frag(&Ps[(wave * 16 + l15) * PLD + kt], sh);
    v16h bf[4];
#pragma unroll
    for (int nt = 0; nt < 4; nt++)
      bf[nt] = ld_frag(&Vt[(nt * 16 + l15) * PLD + kt], sh);
#pragma unroll
    for (int nt = 0; nt < 4; nt++) acc2[nt] = wmma16(af, bf[nt], acc2[nt]);
  }
#pragma unroll
  for (int nt = 0; nt < 4; nt++)
#pragma unroll
    for (int r = 0; r < 8; r++) {
      int m = wave * 16 + sh * 8 + r;
      int n = nt * 16 + l15;
      o[(rowbase + m) * DD + hh * DHH + n] = acc2[nt][r];
    }
}

// ------- final rmsnorm of last token + gate-weighted expert mix --------------
__global__ __launch_bounds__(256)
void mix_kernel(const float* __restrict__ x, const float* __restrict__ lnf,
                const float* __restrict__ gate, float* __restrict__ hmix) {
  __shared__ float red[256];
  int b = blockIdx.x, tid = threadIdx.x;
  float acc[3] = {0.0f, 0.0f, 0.0f};     // DD/256 = 3 elements per thread
  for (int e = 0; e < NE; e++) {
    const float* xr = x + (((size_t)(e * BB + b)) * SS + (SS - 1)) * DD;
    float ss = 0.0f;
    for (int d = tid; d < DD; d += 256) { float vv = xr[d]; ss += vv * vv; }
    red[tid] = ss; __syncthreads();
    for (int s = 128; s > 0; s >>= 1) {
      if (tid < s) red[tid] += red[tid + s];
      __syncthreads();
    }
    float inv = rsqrtf(red[0] / (float)DD + 1e-6f);
    float g = gate[b * NE + e];
    int i = 0;
    for (int d = tid; d < DD; d += 256, i++) acc[i] += g * xr[d] * inv * lnf[d];
    __syncthreads();
  }
  int i = 0;
  for (int d = tid; d < DD; d += 256, i++) hmix[(size_t)b * DD + d] = acc[i];
}

// ------- vocab projection: out[b,n] = hmix[b] . out_W[:,n] + out_b[n] --------
__global__ __launch_bounds__(256)
void out_kernel(const float* __restrict__ hmix, const float* __restrict__ oW,
                const float* __restrict__ ob, float* __restrict__ out) {
  __shared__ float hm[BB * DD];
  int tid = threadIdx.x;
  for (int i = tid; i < BB * DD; i += 256) hm[i] = hmix[i];
  __syncthreads();
  int n = blockIdx.x * 256 + tid;
  if (n >= NOUT) return;
  float a0 = ob[n], a1 = ob[n], a2 = ob[n], a3 = ob[n];
  for (int kk = 0; kk < DD; kk++) {
    float w = oW[(size_t)kk * NOUT + n];   // coalesced across n
    a0 += hm[0 * DD + kk] * w;
    a1 += hm[1 * DD + kk] * w;
    a2 += hm[2 * DD + kk] * w;
    a3 += hm[3 * DD + kk] * w;
  }
  out[0 * (size_t)NOUT + n] = a0;
  out[1 * (size_t)NOUT + n] = a1;
  out[2 * (size_t)NOUT + n] = a2;
  out[3 * (size_t)NOUT + n] = a3;
}

// ---------------------------- driver -----------------------------------------
extern "C" void kernel_launch(void* const* d_in, const int* in_sizes, int n_in,
                              void* d_out, int out_size, void* d_ws, size_t ws_size,
                              hipStream_t stream) {
  (void)in_sizes; (void)n_in; (void)out_size; (void)ws_size;
  const int*   item_seq  = (const int*)d_in[0];
  const int*   item_len  = (const int*)d_in[1];
  const int*   input_ids = (const int*)d_in[2];
  const int*   amask     = (const int*)d_in[3];
  const float* token_emb = (const float*)d_in[4];
  const float* ln1       = (const float*)d_in[5];
  const float* ln2       = (const float*)d_in[6];
  const float* lnf       = (const float*)d_in[7];
  const float* Wq        = (const float*)d_in[8];
  const float* Wk        = (const float*)d_in[9];
  const float* Wv        = (const float*)d_in[10];
  const float* Wo        = (const float*)d_in[11];
  const float* W1        = (const float*)d_in[12];
  const float* W2        = (const float*)d_in[13];
  const float* Aq        = (const float*)d_in[14];
  const float* Bq        = (const float*)d_in[15];
  const float* Av        = (const float*)d_in[16];
  const float* Bv        = (const float*)d_in[17];
  const float* item_emb  = (const float*)d_in[18];
  const float* proj_W    = (const float*)d_in[19];
  const float* proj_b    = (const float*)d_in[20];
  const float* gate_W1   = (const float*)d_in[21];
  const float* gate_b1   = (const float*)d_in[22];
  const float* gate_W2   = (const float*)d_in[23];
  const float* gate_b2   = (const float*)d_in[24];
  const float* out_W     = (const float*)d_in[25];
  const float* out_b     = (const float*)d_in[26];
  float* out = (float*)d_out;

  float* ws   = (float*)d_ws;
  float* gate = ws;                                  // 16
  float* hmix = ws + 64;                             // 3072
  float* x    = ws + 4096;                           // [MROWS, D] f32
  float* h    = x    + (size_t)MROWS * DD;           // [MROWS, D] f32
  float* oatt = h    + (size_t)MROWS * DD;           // [MROWS, D] f32
  float* mid  = oatt + (size_t)MROWS * DD;           // [MROWS, DFF] f32
  _Float16* q16 = (_Float16*)(mid + (size_t)MROWS * DFF);   // [MROWS, D] f16
  _Float16* k16 = q16 + (size_t)MROWS * DD;
  _Float16* v16 = k16 + (size_t)MROWS * DD;

  gate_kernel<<<1, 256, 0, stream>>>(item_seq, item_len, item_emb, proj_W, proj_b,
                                     gate_W1, gate_b1, gate_W2, gate_b2, gate);
  embed_kernel<<<MROWS, 256, 0, stream>>>(input_ids, token_emb, x);

  dim3 g768(DD / 128, MROWS / 128);      // (6,16)
  dim3 g3072(DFF / 128, MROWS / 128);    // (24,16)
  for (int l = 0; l < NL; l++) {
    rmsnorm_kernel<<<MROWS, 256, 0, stream>>>(x, ln1 + (size_t)l * DD, h);
    gemm_kernel<<<g768, 256, 0, stream>>>(h, Wq + (size_t)l * DD * DD, nullptr, q16, MROWS, DD, DD, 4);
    gemm_kernel<<<g768, 256, 0, stream>>>(h, Wk + (size_t)l * DD * DD, nullptr, k16, MROWS, DD, DD, 4);
    gemm_kernel<<<g768, 256, 0, stream>>>(h, Wv + (size_t)l * DD * DD, nullptr, v16, MROWS, DD, DD, 4);
    lora_kernel<<<MROWS / 256, 256, 0, stream>>>(h, Aq, Bq, q16, l);
    lora_kernel<<<MROWS / 256, 256, 0, stream>>>(h, Av, Bv, v16, l);
    attn_kernel<<<NE * BB * NH, 256, 0, stream>>>(q16, k16, v16, amask, oatt);
    gemm_kernel<<<g768, 256, 0, stream>>>(oatt, Wo + (size_t)l * DD * DD, x, nullptr, MROWS, DD, DD, 2);
    rmsnorm_kernel<<<MROWS, 256, 0, stream>>>(x, ln2 + (size_t)l * DD, h);
    gemm_kernel<<<g3072, 256, 0, stream>>>(h, W1 + (size_t)l * DD * DFF, mid, nullptr, MROWS, DFF, DD, 1);
    gemm_kernel<<<g768, 256, 0, stream>>>(mid, W2 + (size_t)l * DFF * DD, x, nullptr, MROWS, DD, DFF, 2);
  }
  mix_kernel<<<BB, 256, 0, stream>>>(x, lnf, gate, hmix);
  out_kernel<<<(NOUT + 255) / 256, 256, 0, stream>>>(hmix, out_W, out_b, out);
}
